// PhraseConductor_36447092474331
// MI455X (gfx1250) — compile-verified, matrix operand order checked by reading, not run on hardware
//
#include <hip/hip_runtime.h>
#include <hip/hip_bf16.h>
#include <math.h>

// ---- problem constants (from reference) ----
#define BB   16
#define SS   8192
#define PP   256
#define DD   256
#define HH   4
#define DHH  64
#define DFF_ 512
#define MM   (BB * PP)   // 4096 rows for all phrase-level GEMMs

typedef __bf16 bf16_t;
typedef __attribute__((ext_vector_type(16))) __bf16 v16bf;
typedef __attribute__((ext_vector_type(8)))  __bf16 v8bf;
typedef __attribute__((ext_vector_type(8)))  float  v8f;

typedef unsigned int tdm_u32x4 __attribute__((ext_vector_type(4)));
typedef int          tdm_i32x8 __attribute__((ext_vector_type(8)));
typedef int          tdm_i32x4 __attribute__((ext_vector_type(4)));

#if defined(__AMDGCN__) && __has_builtin(__builtin_amdgcn_tensor_load_to_lds)
#define USE_TDM 1
#else
#define USE_TDM 0
#endif

// bf16 16x32 A/B operand layout (ISA 7.12.2): per lane, element e holds
// K = (e&7) + 16*(e>>3) + 8*(lane>=16) => two contiguous 16B runs in a
// [row][k] LDS tile -> two ds_load_b128 per fragment.
static __device__ __forceinline__ v16bf frag_from_lds(const __bf16* rowp, int half) {
  v8bf lo = *(const v8bf*)(rowp + 8 * half);
  v8bf hi = *(const v8bf*)(rowp + 16 + 8 * half);
  v16bf f;
#pragma unroll
  for (int e = 0; e < 8; ++e) { f[e] = lo[e]; f[e + 8] = hi[e]; }
  return f;
}

static __device__ __forceinline__ void wmma_step(const __bf16* At, const __bf16* Bt,
                                                 int wm, int wn, int r, int half,
                                                 v8f& acc0, v8f& acc1) {
  v16bf af  = frag_from_lds(&At[(wm * 16 + r) * 32], half);
  v16bf bf0 = frag_from_lds(&Bt[((wn + 0) * 16 + r) * 32], half);
  v16bf bf1 = frag_from_lds(&Bt[((wn + 1) * 16 + r) * 32], half);
  acc0 = __builtin_amdgcn_wmma_f32_16x16x32_bf16(false, af, false, bf0,
                                                 (short)0, acc0, false, false);
  acc1 = __builtin_amdgcn_wmma_f32_16x16x32_bf16(false, af, false, bf1,
                                                 (short)0, acc1, false, false);
}

static __device__ __forceinline__ v8bf cvt8(float4 a, float4 b) {
  v8bf r;
  r[0] = (__bf16)a.x; r[1] = (__bf16)a.y; r[2] = (__bf16)a.z; r[3] = (__bf16)a.w;
  r[4] = (__bf16)b.x; r[5] = (__bf16)b.y; r[6] = (__bf16)b.z; r[7] = (__bf16)b.w;
  return r;
}

#if USE_TDM
// Issue one TDM 2D tile load: tile_dim0=32 (k), tile_dim1=64 (rows), bf16 data.
// D# layout per CDNA5 ISA.
static __device__ __forceinline__ void tdm_load_2d(const __bf16* gptr,
                                                   unsigned lds_off,
                                                   int tensorK, int tensorRows,
                                                   int strideElems) {
  unsigned long long ga = (unsigned long long)(unsigned long long)gptr;
  tdm_u32x4 g0;
  g0[0] = 1u;                                          // count=1 (valid, user)
  g0[1] = lds_off;                                     // lds_addr (bytes)
  g0[2] = (unsigned)(ga & 0xFFFFFFFFull);              // global_addr[31:0]
  g0[3] = (unsigned)((ga >> 32) & 0x01FFFFFFull)       // global_addr[56:32]
        | 0x80000000u;                                 // type=2 ("image")
  tdm_i32x8 g1;
  g1[0] = 1 << 16;                                     // data_size=1 (2 bytes)
  g1[1] = (tensorK & 0xFFFF) << 16;                    // tensor_dim0 lo16
  g1[2] = ((tensorK >> 16) & 0xFFFF)                   // tensor_dim0 hi16
        | ((tensorRows & 0xFFFF) << 16);               // tensor_dim1 lo16
  g1[3] = ((tensorRows >> 16) & 0xFFFF)                // tensor_dim1 hi16
        | (32 << 16);                                  // tile_dim0 = 32
  g1[4] = 64;                                          // tile_dim1=64, tile_dim2=0
  g1[5] = strideElems;                                 // tensor_dim0_stride lo32
  g1[6] = 0;                                           // stride0 hi16 | stride1 lo16
  g1[7] = 0;                                           // stride1 hi32
  tdm_i32x4 z4 = {0, 0, 0, 0};
#if __clang_major__ >= 23
  tdm_i32x8 z8 = {0, 0, 0, 0, 0, 0, 0, 0};
  __builtin_amdgcn_tensor_load_to_lds(g0, g1, z4, z4, z8, 0);
#else
  __builtin_amdgcn_tensor_load_to_lds(g0, g1, z4, z4, 0);
#endif
}
#endif

// ------------------------------------------------------------------
// 1) pooling: zero, atomic scatter, finalize (+sinusoid PE, +mask)
// ------------------------------------------------------------------
__global__ void k_zero(float* __restrict__ p, long n) {
  long i = (long)blockIdx.x * blockDim.x + threadIdx.x;
  if (i < n) p[i] = 0.0f;
}

__global__ void k_cvt_bf16(const float* __restrict__ in, bf16_t* __restrict__ out, long n) {
  long i = (long)blockIdx.x * blockDim.x + threadIdx.x;
  if (i < n) out[i] = (__bf16)in[i];
}

__global__ void k_pool_scatter(const float* __restrict__ tok,
                               const int* __restrict__ pid,
                               const unsigned char* __restrict__ amask,
                               float* __restrict__ sums,
                               float* __restrict__ cnt) {
  long i = (long)blockIdx.x * blockDim.x + threadIdx.x;   // over B*S*D
  if (i >= (long)BB * SS * DD) return;
  int d = (int)(i % DD);
  long bs = i / DD;
  int b = (int)(bs / SS), s = (int)(bs % SS);
  if (!amask[(long)b * SS + s]) return;
  int ph = pid[(long)b * SS + s];
  atomicAdd(&sums[((long)b * PP + ph) * DD + d], tok[i]);
  if (d == 0) atomicAdd(&cnt[b * PP + ph], 1.0f);
}

__global__ void k_finalize_pool(const float* __restrict__ sums,
                                const float* __restrict__ cnt,
                                const unsigned char* __restrict__ pmask,
                                float* __restrict__ x) {
  long i = (long)blockIdx.x * blockDim.x + threadIdx.x;   // over M*D
  if (i >= (long)MM * DD) return;
  int d  = (int)(i % DD);
  int bp = (int)(i / DD);
  int p  = bp % PP;
  float c = cnt[bp];
  float v = (c > 0.0f) ? sums[i] / fmaxf(c, 1.0f) : 0.0f;
  if (!pmask[bp]) v = 0.0f;
  int pair = d >> 1;
  float div = expf(-logf(10000.0f) * (float)(2 * pair) / (float)DD);
  float pe  = (d & 1) ? cosf((float)p * div) : sinf((float)p * div);
  x[i] = v + pe;
}

// ------------------------------------------------------------------
// 2) block-tiled bf16 WMMA GEMM with TDM double-buffered staging.
//    C = act(A[M,K]bf16 @ W[N,K]bf16^T + bias [+resid]) [*mask]
//    256 threads = 8 waves; 64x64 C tile; K stepped by 32.
// ------------------------------------------------------------------
template <int GELU, int RESID, int MASK>
__global__ void k_gemm_t(const bf16_t* __restrict__ A, int lda,
                         const bf16_t* __restrict__ W, int ldw,
                         const float* __restrict__ bias,
                         const float* __restrict__ resid, int ldr,
                         const unsigned char* __restrict__ pmask,
                         float* __restrict__ C, int ldc,
                         int K, int tilesN) {
  __shared__ __bf16 Alds[2][64 * 32];
  __shared__ __bf16 Blds[2][64 * 32];
  int bm = blockIdx.x / tilesN, bn = blockIdx.x - bm * tilesN;
  int t = threadIdx.x;
  int lane = t & 31;
  int half = lane >> 4, r = lane & 15;
  int wm = (t >> 5) & 3;
  int wn = (t >> 7) * 2;
  const bf16_t* Ag = A + (long)(bm * 64) * lda;
  const bf16_t* Wg = W + (long)(bn * 64) * ldw;
  v8f acc0 = {}, acc1 = {};
#if USE_TDM
  bool issuer = (t < 32);
  if (issuer) {
    tdm_load_2d(Ag, (unsigned)(unsigned long long)&Alds[0][0], K, 64, lda);
    tdm_load_2d(Wg, (unsigned)(unsigned long long)&Blds[0][0], K, 64, ldw);
  }
  int buf = 0;
  for (int k0 = 0; k0 < K; k0 += 32) {
    if (issuer) __builtin_amdgcn_s_wait_tensorcnt(0);   // buf ready
    __syncthreads();                                    // release readers
    if (issuer && (k0 + 32 < K)) {                      // prefetch next slab
      tdm_load_2d(Ag + (k0 + 32), (unsigned)(unsigned long long)&Alds[buf ^ 1][0], K, 64, lda);
      tdm_load_2d(Wg + (k0 + 32), (unsigned)(unsigned long long)&Blds[buf ^ 1][0], K, 64, ldw);
    }
    wmma_step(Alds[buf], Blds[buf], wm, wn, r, half, acc0, acc1);
    __syncthreads();                                    // done reading buf
    buf ^= 1;
  }
#else
  int srow = t >> 2;           // 0..63
  int skg  = (t & 3) << 3;     // 0,8,16,24
  for (int k0 = 0; k0 < K; k0 += 32) {
    v8bf av = *(const v8bf*)(Ag + (long)srow * lda + k0 + skg);
    v8bf bv = *(const v8bf*)(Wg + (long)srow * ldw + k0 + skg);
    __syncthreads();
    *(v8bf*)&Alds[0][srow * 32 + skg] = av;
    *(v8bf*)&Blds[0][srow * 32 + skg] = bv;
    __syncthreads();
    wmma_step(Alds[0], Blds[0], wm, wn, r, half, acc0, acc1);
  }
#endif
#pragma unroll
  for (int sub = 0; sub < 2; ++sub) {
    v8f acc = sub ? acc1 : acc0;
    int col = bn * 64 + (wn + sub) * 16 + r;
    float bs = bias ? bias[col] : 0.0f;
#pragma unroll
    for (int rr = 0; rr < 8; ++rr) {
      int row = bm * 64 + wm * 16 + rr + (half << 3);
      float v = acc[rr] + bs;
      if (RESID) v += resid[(long)row * ldr + col];
      if (GELU)  v = 0.5f * v * (1.0f + erff(v * 0.70710678118f));
      if (MASK)  v = pmask[row] ? v : 0.0f;
      C[(long)row * ldc + col] = v;
    }
  }
}

// simple per-wave GEMM for the tiny head projections (N in {16,32,64})
__global__ void k_gemm_wave(const float* __restrict__ A,
                            const float* __restrict__ W,
                            const float* __restrict__ bias,
                            float* __restrict__ C,
                            int N, int K) {
  int wave = blockIdx.x * (blockDim.x >> 5) + (threadIdx.x >> 5);
  int lane = threadIdx.x & 31;
  int tilesN = N >> 4;
  if (wave >= (MM >> 4) * tilesN) return;
  int tm = wave / tilesN, tn = wave - tm * tilesN;
  int half = lane >> 4, r = lane & 15;
  const float* Ab = A + (long)(tm * 16 + r) * K;
  const float* Wb = W + (long)(tn * 16 + r) * K;
  v8f acc = {};
  for (int k0 = 0; k0 < K; k0 += 32) {
    v16bf av, bv;
#pragma unroll
    for (int e = 0; e < 16; ++e) {
      int kk = k0 + (e & 7) + ((e >> 3) << 4) + (half << 3);
      av[e] = (__bf16)Ab[kk];
      bv[e] = (__bf16)Wb[kk];
    }
    acc = __builtin_amdgcn_wmma_f32_16x16x32_bf16(false, av, false, bv,
                                                  (short)0, acc, false, false);
  }
  int col = tn * 16 + r;
  float bs = bias[col];
#pragma unroll
  for (int rr = 0; rr < 8; ++rr) {
    int row = tm * 16 + rr + (half << 3);
    C[(long)row * N + col] = acc[rr] + bs;
  }
}

// ------------------------------------------------------------------
// 3) attention: tiled QK^T (+scale +key bias), softmax, tiled P.V
// ------------------------------------------------------------------
__global__ void k_attn_scores(const float* __restrict__ qkv,
                              const unsigned char* __restrict__ pmask,
                              float* __restrict__ scores, float scale) {
  __shared__ __bf16 Alds[64 * 32];
  __shared__ __bf16 Blds[64 * 32];
  int bh = blockIdx.y;
  int b = bh / HH, h = bh - b * HH;
  int bm = blockIdx.x >> 2, bn = blockIdx.x & 3;
  int t = threadIdx.x;
  int lane = t & 31;
  int half = lane >> 4, r = lane & 15;
  int wm = (t >> 5) & 3;
  int wn = (t >> 7) * 2;
  int srow = t >> 2;
  int skg  = (t & 3) << 3;
  const float* Qg = qkv + ((long)b * PP + bm * 64 + srow) * (3 * DD) + h * DHH + skg;
  const float* Kg = qkv + ((long)b * PP + bn * 64 + srow) * (3 * DD) + DD + h * DHH + skg;
  v8f acc0 = {}, acc1 = {};
  for (int k0 = 0; k0 < DHH; k0 += 32) {
    float4 a0 = *(const float4*)(Qg + k0);
    float4 a1 = *(const float4*)(Qg + k0 + 4);
    float4 b0 = *(const float4*)(Kg + k0);
    float4 b1 = *(const float4*)(Kg + k0 + 4);
    __syncthreads();
    *(v8bf*)&Alds[srow * 32 + skg] = cvt8(a0, a1);
    *(v8bf*)&Blds[srow * 32 + skg] = cvt8(b0, b1);
    __syncthreads();
    wmma_step(Alds, Blds, wm, wn, r, half, acc0, acc1);
  }
#pragma unroll
  for (int sub = 0; sub < 2; ++sub) {
    v8f acc = sub ? acc1 : acc0;
    int col = bn * 64 + (wn + sub) * 16 + r;
    float kb = pmask[b * PP + col] ? 0.0f : -1e9f;
#pragma unroll
    for (int rr = 0; rr < 8; ++rr) {
      int row = bm * 64 + wm * 16 + rr + (half << 3);
      scores[((long)bh * PP + row) * PP + col] = acc[rr] * scale + kb;
    }
  }
}

__global__ void k_softmax256(float* __restrict__ xrow) {
  __shared__ float sh[256];
  int row = blockIdx.x, t = threadIdx.x;
  float v = xrow[(long)row * 256 + t];
  sh[t] = v; __syncthreads();
  for (int s = 128; s > 0; s >>= 1) { if (t < s) sh[t] = fmaxf(sh[t], sh[t + s]); __syncthreads(); }
  float m = sh[0]; __syncthreads();
  float e = expf(v - m);
  sh[t] = e; __syncthreads();
  for (int s = 128; s > 0; s >>= 1) { if (t < s) sh[t] += sh[t + s]; __syncthreads(); }
  xrow[(long)row * 256 + t] = e / sh[0];
}

__global__ void k_attn_av(const float* __restrict__ probs,
                          const float* __restrict__ qkv,
                          float* __restrict__ o) {
  __shared__ __bf16 Alds[64 * 32];
  __shared__ __bf16 Blds[64 * 32];   // [col(dh)][k] -- V staged K-major
  int bh = blockIdx.y;
  int b = bh / HH, h = bh - b * HH;
  int bm = blockIdx.x;
  int t = threadIdx.x;
  int lane = t & 31;
  int half = lane >> 4, r = lane & 15;
  int wm = (t >> 5) & 3;
  int wn = (t >> 7) * 2;
  int srow = t >> 2;
  int skg  = (t & 3) << 3;
  const float* Ag = probs + ((long)bh * PP + bm * 64 + srow) * PP + skg;
  const float* Vg = qkv + (long)b * PP * (3 * DD) + 2 * DD + h * DHH + srow;
  v8f acc0 = {}, acc1 = {};
  for (int k0 = 0; k0 < PP; k0 += 32) {
    float4 a0 = *(const float4*)(Ag + k0);
    float4 a1 = *(const float4*)(Ag + k0 + 4);
    v8bf bpk;
#pragma unroll
    for (int j = 0; j < 8; ++j)
      bpk[j] = (__bf16)Vg[(long)(k0 + skg + j) * (3 * DD)];
    __syncthreads();
    *(v8bf*)&Alds[srow * 32 + skg] = cvt8(a0, a1);
    *(v8bf*)&Blds[srow * 32 + skg] = bpk;
    __syncthreads();
    wmma_step(Alds, Blds, wm, wn, r, half, acc0, acc1);
  }
#pragma unroll
  for (int sub = 0; sub < 2; ++sub) {
    v8f acc = sub ? acc1 : acc0;
    int col = (wn + sub) * 16 + r;
#pragma unroll
    for (int rr = 0; rr < 8; ++rr) {
      int row = bm * 64 + wm * 16 + rr + (half << 3);
      o[((long)b * PP + row) * DD + h * DHH + col] = acc[rr];
    }
  }
}

// ------------------------------------------------------------------
// 4) layernorm, head softmax, probs @ E
// ------------------------------------------------------------------
__global__ void k_layernorm(const float* __restrict__ in,
                            const float* __restrict__ g,
                            const float* __restrict__ b,
                            float* __restrict__ out) {
  __shared__ float sh[256];
  int row = blockIdx.x, t = threadIdx.x;
  float v = in[(long)row * 256 + t];
  sh[t] = v; __syncthreads();
  for (int s = 128; s > 0; s >>= 1) { if (t < s) sh[t] += sh[t + s]; __syncthreads(); }
  float mean = sh[0] * (1.0f / 256.0f); __syncthreads();
  float dv = v - mean;
  sh[t] = dv * dv; __syncthreads();
  for (int s = 128; s > 0; s >>= 1) { if (t < s) sh[t] += sh[t + s]; __syncthreads(); }
  float var = sh[0] * (1.0f / 256.0f);
  out[(long)row * 256 + t] = dv * rsqrtf(var + 1e-5f) * g[t] + b[t];
}

__global__ void k_softmax_small(const float* __restrict__ logits,
                                float* __restrict__ probs, int V) {
  int m = blockIdx.x * blockDim.x + threadIdx.x;
  if (m >= MM) return;
  float mx = -1e30f;
  for (int v = 0; v < V; ++v) mx = fmaxf(mx, logits[(long)m * V + v]);
  float sum = 0.0f;
  for (int v = 0; v < V; ++v) sum += expf(logits[(long)m * V + v] - mx);
  float inv = 1.0f / sum;
  for (int v = 0; v < V; ++v) probs[(long)m * V + v] = expf(logits[(long)m * V + v] - mx) * inv;
}

__global__ void k_probsE(const float* __restrict__ probs,
                         const float* __restrict__ E,
                         float* __restrict__ feats, int V, int colOff) {
  long i = (long)blockIdx.x * blockDim.x + threadIdx.x;
  if (i >= (long)MM * DD) return;
  int d = (int)(i % DD), m = (int)(i / DD);
  float s = 0.0f;
  for (int v = 0; v < V; ++v) s += probs[(long)m * V + v] * E[(long)v * DD + d];
  feats[(long)m * (4 * DD) + colOff + d] = s;
}

// ------------------------------------------------------------------
// host orchestration
// ------------------------------------------------------------------
extern "C" void kernel_launch(void* const* d_in, const int* in_sizes, int n_in,
                              void* d_out, int out_size, void* d_ws, size_t ws_size,
                              hipStream_t stream) {
  const float*         tok   = (const float*)d_in[0];
  const int*           pid   = (const int*)d_in[1];
  const unsigned char* amask = (const unsigned char*)d_in[2];
  const unsigned char* pmask = (const unsigned char*)d_in[3];
  const float* Wc   = (const float*)d_in[4];
  const float* bc   = (const float*)d_in[5];
  const float* fn_b = (const float*)d_in[6];
  const float* fn_g = (const float*)d_in[7];
  const float* E_art = (const float*)d_in[8],  *W_art = (const float*)d_in[9],  *b_art = (const float*)d_in[10];
  const float* E_den = (const float*)d_in[11], *W_den = (const float*)d_in[12], *b_den = (const float*)d_in[13];
  const float* E_dyn = (const float*)d_in[14], *W_dyn = (const float*)d_in[15], *b_dyn = (const float*)d_in[16];
  const float* E_tem = (const float*)d_in[17], *W_tem = (const float*)d_in[18], *b_tem = (const float*)d_in[19];
  const int IN_L0 = 20, IN_L1 = 32;

  float* ws    = (float*)d_ws;
  float* sums  = ws;  ws += (long)MM * DD;
  float* cnt   = ws;  ws += MM;
  float* x     = ws;  ws += (long)MM * DD;
  float* y     = ws;  ws += (long)MM * DD;
  float* qkv   = ws;  ws += (long)MM * 3 * DD;
  float* o     = ws;  ws += (long)MM * DD;
  float* sc    = ws;  ws += (long)BB * HH * PP * PP;
  float* h1    = ws;  ws += (long)MM * DFF_;
  float* probs = ws;  ws += (long)MM * 144;
  float* feats = ws;  ws += (long)MM * 4 * DD;
  // bf16 mirrors
  bf16_t* bws    = (bf16_t*)ws;
  bf16_t* xb     = bws;  bws += (long)MM * DD;
  bf16_t* ob     = bws;  bws += (long)MM * DD;
  bf16_t* h1b    = bws;  bws += (long)MM * DFF_;
  bf16_t* featsb = bws;  bws += (long)MM * 4 * DD;
  bf16_t* Wqkvb  = bws;  bws += (long)768 * DD;
  bf16_t* Wob    = bws;  bws += (long)DD * DD;
  bf16_t* W1b    = bws;  bws += (long)DFF_ * DD;
  bf16_t* W2b    = bws;  bws += (long)DD * DFF_;
  bf16_t* Wcb    = bws;  bws += (long)DD * 4 * DD;

  float* out    = (float*)d_out;
  float* ph     = out;
  float* cs     = out + (long)MM * DD;
  float* lg     = out + 2l * MM * DD;
  float* lg_dyn = lg;
  float* lg_tem = lg + (long)MM * 32;
  float* lg_art = lg + (long)MM * 96;
  float* lg_den = lg + (long)MM * 112;

  auto cvt = [&](const float* src, bf16_t* dst, long n) {
    k_cvt_bf16<<<(int)((n + 255) / 256), 256, 0, stream>>>(src, dst, n);
  };

  // ---- pooling ----
  {
    long nz = (long)MM * DD + MM;
    k_zero<<<(int)((nz + 255) / 256), 256, 0, stream>>>(sums, nz);
    long nt = (long)BB * SS * DD;
    k_pool_scatter<<<(int)((nt + 255) / 256), 256, 0, stream>>>(tok, pid, amask, sums, cnt);
    long nf = (long)MM * DD;
    k_finalize_pool<<<(int)((nf + 255) / 256), 256, 0, stream>>>(sums, cnt, pmask, x);
    cvt(x, xb, (long)MM * DD);
  }

  const float scale = 0.125f;  // 1/sqrt(DH)
  for (int l = 0; l < 2; ++l) {
    int base = (l == 0) ? IN_L0 : IN_L1;
    const float* W1   = (const float*)d_in[base + 0];
    const float* W2   = (const float*)d_in[base + 1];
    const float* Wo   = (const float*)d_in[base + 2];
    const float* Wqkv = (const float*)d_in[base + 3];
    const float* b1   = (const float*)d_in[base + 4];
    const float* b2   = (const float*)d_in[base + 5];
    const float* be1  = (const float*)d_in[base + 6];
    const float* be2  = (const float*)d_in[base + 7];
    const float* bo   = (const float*)d_in[base + 8];
    const float* bqkv = (const float*)d_in[base + 9];
    const float* g1   = (const float*)d_in[base + 10];
    const float* g2   = (const float*)d_in[base + 11];

    cvt(Wqkv, Wqkvb, (long)768 * DD);
    cvt(Wo,   Wob,   (long)DD * DD);
    cvt(W1,   W1b,   (long)DFF_ * DD);
    cvt(W2,   W2b,   (long)DD * DFF_);

    // qkv = x @ Wqkv^T + bqkv
    k_gemm_t<0,0,0><<<dim3((MM/64) * (768/64)), 256, 0, stream>>>(
        xb, DD, Wqkvb, DD, bqkv, nullptr, 0, nullptr, qkv, 3*DD, DD, 768/64);

    k_attn_scores<<<dim3(16, BB*HH), 256, 0, stream>>>(qkv, pmask, sc, scale);
    k_softmax256<<<BB*HH*PP, 256, 0, stream>>>(sc);
    k_attn_av<<<dim3(4, BB*HH), 256, 0, stream>>>(sc, qkv, o);
    cvt(o, ob, (long)MM * DD);

    // y = o @ Wo^T + bo + x ; x = LN(y)
    k_gemm_t<0,1,0><<<dim3((MM/64) * (DD/64)), 256, 0, stream>>>(
        ob, DD, Wob, DD, bo, x, DD, nullptr, y, DD, DD, DD/64);
    k_layernorm<<<MM, 256, 0, stream>>>(y, g1, be1, x);
    cvt(x, xb, (long)MM * DD);

    // h1 = gelu(x @ W1^T + b1)
    k_gemm_t<1,0,0><<<dim3((MM/64) * (DFF_/64)), 256, 0, stream>>>(
        xb, DD, W1b, DD, b1, nullptr, 0, nullptr, h1, DFF_, DD, DFF_/64);
    cvt(h1, h1b, (long)MM * DFF_);
    // y = h1 @ W2^T + b2 + x ; x = LN(y)
    k_gemm_t<0,1,0><<<dim3((MM/64) * (DD/64)), 256, 0, stream>>>(
        h1b, DFF_, W2b, DFF_, b2, x, DD, nullptr, y, DD, DFF_, DD/64);
    k_layernorm<<<MM, 256, 0, stream>>>(y, g2, be2, x);
    cvt(x, xb, (long)MM * DD);
  }

  k_layernorm<<<MM, 256, 0, stream>>>(x, fn_g, fn_b, ph);

  // heads (output order: dynamics, tempo, articulation, density)
  struct HeadDesc { const float *W, *b, *E; int V; float* logit; float* prob; int colOff; };
  HeadDesc heads[4] = {
    { W_dyn, b_dyn, E_dyn, 32, lg_dyn, probs,                 0   },
    { W_tem, b_tem, E_tem, 64, lg_tem, probs + (long)MM * 32, 256 },
    { W_art, b_art, E_art, 16, lg_art, probs + (long)MM * 96, 512 },
    { W_den, b_den, E_den, 32, lg_den, probs + (long)MM * 112,768 },
  };
  for (int j = 0; j < 4; ++j) {
    HeadDesc& hd = heads[j];
    int tiles = (MM / 16) * (hd.V / 16);
    k_gemm_wave<<<(tiles + 3) / 4, 128, 0, stream>>>(ph, hd.W, hd.b, hd.logit, hd.V, DD);
    k_softmax_small<<<(MM + 255) / 256, 256, 0, stream>>>(hd.logit, hd.prob, hd.V);
    long np = (long)MM * DD;
    k_probsE<<<(int)((np + 255) / 256), 256, 0, stream>>>(hd.prob, hd.E, feats, hd.V, hd.colOff);
  }

  // cs = (ph + feats @ Wc^T + bc) * phrase_mask
  cvt(feats, featsb, (long)MM * 4 * DD);
  cvt(Wc, Wcb, (long)DD * 4 * DD);
  k_gemm_t<0,1,1><<<dim3((MM/64) * (DD/64)), 256, 0, stream>>>(
      featsb, 4*DD, Wcb, 4*DD, bc, ph, DD, pmask, cs, DD, 4*DD, DD/64);
}
